// TensorNet_Ext_65060164599842
// MI455X (gfx1250) — matrix operand order repaired and checked by reading, not compiled, not run on hardware
//
#include <hip/hip_runtime.h>
#include <hip/hip_bf16.h>
#include <math.h>

#define NN 10000
#define EE 160000
#define CC 64
#define RR 32
#define NC (NN * CC)   // 640000 (n,c) pairs, one compact plane

typedef __attribute__((ext_vector_type(16))) _Float16 v16h;
typedef __attribute__((ext_vector_type(8)))  float    v8f;

__device__ __forceinline__ v8f wmma_f16(v16h a, v16h b, v8f c) {
  // D(16x16,f32) = A(16x32,f16) * B(32x16,f16) + C
  return __builtin_amdgcn_wmma_f32_16x16x32_f16(false, a, false, b, (short)0, c, false, false);
}

// Fast silu using hardware TRANS ops (v_exp_f32 + v_rcp_f32); these co-execute
// with WMMA on CDNA5, keeping the VALU tail off the matrix-core critical path.
__device__ __forceinline__ float silu_f(float x) {
  return x * __builtin_amdgcn_rcpf(1.0f + __expf(-x));
}

// A-fragment K index for 16-bit 16x32 A tiles (lane&15 = row M, lane>>4 = half select)
__device__ __forceinline__ int a_kidx(int h, int hl) {
  return ((h < 8) ? h : (h + 8)) + 8 * hl;   // K in [0,32)
}

// ---------------------------------------------------------------------------
// Kernel 0: convert all weights to f16 into one contiguous workspace region.
// Layout: W1h[32*64] | W2h[64*128] | W3h[128*192] | WtH[6*64*64]
// ---------------------------------------------------------------------------
__global__ void __launch_bounds__(256) convert_weights_kernel(
    const float* __restrict__ W1, const float* __restrict__ W2,
    const float* __restrict__ W3, const float* __restrict__ Wt,
    _Float16* __restrict__ out)
{
  int i = blockIdx.x * 256 + threadIdx.x;
  if (i < 2048)        out[i] = (_Float16)W1[i];
  else if (i < 10240)  out[i] = (_Float16)W2[i - 2048];
  else if (i < 34816)  out[i] = (_Float16)W3[i - 10240];
  else if (i < 59392)  out[i] = (_Float16)Wt[i - 34816];
}

// ---------------------------------------------------------------------------
// Kernel 1: edge scalar MLP, fully WMMA.  One wave owns a 16-edge tile.
// h1 = silu(ea@W1+b1) [16x64]; h2 = silu(h1@W2+b2) [16x128];
// a  = silu(h2@W3+b3) * cutoff(w)  -> afac[E][192] (f16)
// Layer outputs are re-laid-out D->A through a wave-private LDS stage.
// ---------------------------------------------------------------------------
__global__ void __launch_bounds__(256) edge_mlp_kernel(
    const float* __restrict__ edge_attr,    // [E,32]
    const float* __restrict__ edge_weight,  // [E]
    const float* __restrict__ b1, const float* __restrict__ b2,
    const float* __restrict__ b3,
    const _Float16* __restrict__ W1h,       // [32][64]
    const _Float16* __restrict__ W2h,       // [64][128]
    const _Float16* __restrict__ W3h,       // [128][192]
    _Float16* __restrict__ afac)            // [E][192]
{
  __shared__ _Float16 stage[8][16 * 128];   // 32 KB, one [16x128] stage per wave
  const int lane = threadIdx.x & 31;
  const int wv   = threadIdx.x >> 5;
  const int hl   = lane >> 4;
  const int nl   = lane & 15;               // A-row / D-column index
  const int e0   = (blockIdx.x * 8 + wv) * 16;
  _Float16* stg  = stage[wv];

  // ---- layer 1: [16x32]@[32x64] ----
  v16h a1;
  {
    const float* row = edge_attr + (size_t)(e0 + nl) * RR;
    #pragma unroll
    for (int h = 0; h < 16; ++h) a1[h] = (_Float16)row[a_kidx(h, hl)];
  }
  #pragma unroll
  for (int nt = 0; nt < 4; ++nt) {
    v16h b;
    #pragma unroll
    for (int h = 0; h < 16; ++h) b[h] = W1h[a_kidx(h, hl) * 64 + nt * 16 + nl];
    v8f acc;
    float bias = b1[nt * 16 + nl];
    #pragma unroll
    for (int r = 0; r < 8; ++r) acc[r] = bias;
    acc = wmma_f16(a1, b, acc);
    #pragma unroll
    for (int r = 0; r < 8; ++r) {
      int m = r + 8 * hl;                       // D row
      stg[m * 128 + nt * 16 + nl] = (_Float16)silu_f(acc[r]);
    }
  }

  // ---- layer 2: [16x64]@[64x128] ----
  v16h a2[2];
  #pragma unroll
  for (int kc = 0; kc < 2; ++kc)
    #pragma unroll
    for (int h = 0; h < 16; ++h)
      a2[kc][h] = stg[nl * 128 + a_kidx(h, hl) + 32 * kc];
  #pragma unroll
  for (int nt = 0; nt < 8; ++nt) {
    v8f acc;
    float bias = b2[nt * 16 + nl];
    #pragma unroll
    for (int r = 0; r < 8; ++r) acc[r] = bias;
    #pragma unroll
    for (int kc = 0; kc < 2; ++kc) {
      v16h b;
      #pragma unroll
      for (int h = 0; h < 16; ++h) b[h] = W2h[(kc * 32 + a_kidx(h, hl)) * 128 + nt * 16 + nl];
      acc = wmma_f16(a2[kc], b, acc);
    }
    #pragma unroll
    for (int r = 0; r < 8; ++r) {
      int m = r + 8 * hl;
      stg[m * 128 + nt * 16 + nl] = (_Float16)silu_f(acc[r]);
    }
  }

  // ---- layer 3: [16x128]@[128x192], fused cosine cutoff ----
  v16h a3[4];
  #pragma unroll
  for (int kc = 0; kc < 4; ++kc)
    #pragma unroll
    for (int h = 0; h < 16; ++h)
      a3[kc][h] = stg[nl * 128 + a_kidx(h, hl) + 32 * kc];

  float cw[8];
  #pragma unroll
  for (int r = 0; r < 8; ++r) {
    float w = edge_weight[e0 + r + 8 * hl];
    float c = 0.5f * (__cosf(w * 0.62831853071795864f) + 1.0f);  // pi/5
    cw[r] = (w < 5.0f) ? c : 0.0f;
  }

  #pragma unroll
  for (int nt = 0; nt < 12; ++nt) {
    v8f acc;
    float bias = b3[nt * 16 + nl];
    #pragma unroll
    for (int r = 0; r < 8; ++r) acc[r] = bias;
    #pragma unroll
    for (int kc = 0; kc < 4; ++kc) {
      v16h b;
      #pragma unroll
      for (int h = 0; h < 16; ++h) b[h] = W3h[(kc * 32 + a_kidx(h, hl)) * 192 + nt * 16 + nl];
      acc = wmma_f16(a3[kc], b, acc);
    }
    #pragma unroll
    for (int r = 0; r < 8; ++r) {
      int m = r + 8 * hl;
      afac[(size_t)(e0 + m) * 192 + nt * 16 + nl] = (_Float16)(silu_f(acc[r]) * cw[r]);
    }
  }
}

// ---------------------------------------------------------------------------
// Kernel 2: per-(n,c) normalize + irreducible decomposition into 9 compact
// planes: [0]=tr/3, [1..3]=a01,a02,a12 (antisym), [4..8]=s00,s01,s02,s11,s12.
// ---------------------------------------------------------------------------
__global__ void __launch_bounds__(256) node_decompose_kernel(
    const float* __restrict__ X, float* __restrict__ Xn, float* __restrict__ planes)
{
  int i = blockIdx.x * 256 + threadIdx.x;
  if (i >= NC) return;
  float t[9], s = 0.0f;
  #pragma unroll
  for (int j = 0; j < 9; ++j) { t[j] = X[(size_t)i * 9 + j]; s += t[j] * t[j]; }
  float inv = 1.0f / (s + 1.0f);
  #pragma unroll
  for (int j = 0; j < 9; ++j) { t[j] *= inv; Xn[(size_t)i * 9 + j] = t[j]; }
  float tr3 = (t[0] + t[4] + t[8]) * (1.0f / 3.0f);
  planes[0 * NC + i] = tr3;
  planes[1 * NC + i] = 0.5f * (t[1] - t[3]);
  planes[2 * NC + i] = 0.5f * (t[2] - t[6]);
  planes[3 * NC + i] = 0.5f * (t[5] - t[7]);
  planes[4 * NC + i] = t[0] - tr3;
  planes[5 * NC + i] = 0.5f * (t[1] + t[3]);
  planes[6 * NC + i] = 0.5f * (t[2] + t[6]);
  planes[7 * NC + i] = t[4] - tr3;
  planes[8 * NC + i] = 0.5f * (t[5] + t[7]);
}

// ---------------------------------------------------------------------------
// Kernel 3: component-mix GEMM (WMMA). For each of 9 compact planes:
// dst[p] = src[p] @ W{0|1|2}  with p=0 -> W0, p in 1..3 -> W1, p in 4..8 -> W2.
// One wave = one 16x16 output tile, K=64 (2 wmmas).
// Units: 9 comps * 625 M-tiles * 4 N-tiles = 22500.
// ---------------------------------------------------------------------------
__global__ void __launch_bounds__(256) mix_gemm_kernel(
    const float* __restrict__ src, const _Float16* __restrict__ Wset,
    float* __restrict__ dst)
{
  const int unit = blockIdx.x * 8 + (threadIdx.x >> 5);
  if (unit >= 9 * 625 * 4) return;              // wave-uniform exit
  const int comp = unit / (625 * 4);
  const int rem  = unit % (625 * 4);
  const int mt = rem >> 2, nt = rem & 3;
  const int lane = threadIdx.x & 31, hl = lane >> 4, nl = lane & 15;
  const int wsel = (comp == 0) ? 0 : ((comp < 4) ? 1 : 2);
  const float* A = src + comp * NC + (mt * 16 + nl) * CC;
  const _Float16* W = Wset + wsel * 4096 + nt * 16;

  v8f acc;
  #pragma unroll
  for (int r = 0; r < 8; ++r) acc[r] = 0.0f;
  #pragma unroll
  for (int kc = 0; kc < 2; ++kc) {
    v16h a, b;
    #pragma unroll
    for (int h = 0; h < 16; ++h) a[h] = (_Float16)A[a_kidx(h, hl) + 32 * kc];
    #pragma unroll
    for (int h = 0; h < 16; ++h) b[h] = W[(kc * 32 + a_kidx(h, hl)) * 64 + nl];
    acc = wmma_f16(a, b, acc);
  }
  float* D = dst + comp * NC + (mt * 16) * CC + nt * 16 + nl;
  #pragma unroll
  for (int r = 0; r < 8; ++r) D[(r + 8 * hl) * CC] = acc[r];
}

// ---------------------------------------------------------------------------
// Kernel 4: message passing on compact planes. Thread = (edge, channel).
// msg[p][src] += factor(p) * mixed[p][dst], 9 float atomics (coalesced in c).
// ---------------------------------------------------------------------------
__global__ void __launch_bounds__(256) edge_scatter_kernel(
    const int* __restrict__ ei, const _Float16* __restrict__ afac,
    const float* __restrict__ mixed, float* __restrict__ msg)
{
  int i = blockIdx.x * 256 + threadIdx.x;
  if (i >= EE * CC) return;
  int e = i >> 6, c = i & 63;
  int s = ei[e], d = ei[EE + e];
  float aI = (float)afac[(size_t)e * 192 + 3 * c + 0];
  float aA = (float)afac[(size_t)e * 192 + 3 * c + 1];
  float aS = (float)afac[(size_t)e * 192 + 3 * c + 2];
  int doff = d * CC + c, soff = s * CC + c;
  atomicAdd(&msg[0 * NC + soff], aI * mixed[0 * NC + doff]);
  #pragma unroll
  for (int p = 1; p < 4; ++p)
    atomicAdd(&msg[p * NC + soff], aA * mixed[p * NC + doff]);
  #pragma unroll
  for (int p = 4; p < 9; ++p)
    atomicAdd(&msg[p * NC + soff], aS * mixed[p * NC + doff]);
}

// ---------------------------------------------------------------------------
// Per-(n,c) 3x3 rebuild from compacts (I + A + S, with s22 = -s00-s11).
// ---------------------------------------------------------------------------
__device__ __forceinline__ void rebuild3x3(const float c[9], float T[9]) {
  float tI = c[0], a01 = c[1], a02 = c[2], a12 = c[3];
  float s00 = c[4], s01 = c[5], s02 = c[6], s11 = c[7], s12 = c[8];
  float s22 = -s00 - s11;
  T[0] = tI + s00;  T[1] =  a01 + s01; T[2] =  a02 + s02;
  T[3] = -a01 + s01; T[4] = tI + s11;  T[5] =  a12 + s12;
  T[6] = -a02 + s02; T[7] = -a12 + s12; T[8] = tI + s22;
}

// ---------------------------------------------------------------------------
// Kernel 5: M = msg@Y + Y@msg, decompose, divide by (||M||^2+1) -> planes2.
// ---------------------------------------------------------------------------
__global__ void __launch_bounds__(256) node_update_kernel(
    const float* __restrict__ mixed, const float* __restrict__ msg,
    float* __restrict__ planes2)
{
  int i = blockIdx.x * 256 + threadIdx.x;
  if (i >= NC) return;
  float cm[9], cy[9];
  #pragma unroll
  for (int p = 0; p < 9; ++p) { cm[p] = msg[p * NC + i]; cy[p] = mixed[p * NC + i]; }
  float Mm[9], Y[9], P[9];
  rebuild3x3(cm, Mm);
  rebuild3x3(cy, Y);
  #pragma unroll
  for (int r = 0; r < 3; ++r)
    #pragma unroll
    for (int c = 0; c < 3; ++c) {
      float v = 0.0f;
      #pragma unroll
      for (int k = 0; k < 3; ++k)
        v += Mm[r * 3 + k] * Y[k * 3 + c] + Y[r * 3 + k] * Mm[k * 3 + c];
      P[r * 3 + c] = v;
    }
  float fr = 0.0f;
  #pragma unroll
  for (int j = 0; j < 9; ++j) fr += P[j] * P[j];
  float inv = 1.0f / (fr + 1.0f);
  float tr3 = (P[0] + P[4] + P[8]) * (1.0f / 3.0f);
  planes2[0 * NC + i] = tr3 * inv;
  planes2[1 * NC + i] = 0.5f * (P[1] - P[3]) * inv;
  planes2[2 * NC + i] = 0.5f * (P[2] - P[6]) * inv;
  planes2[3 * NC + i] = 0.5f * (P[5] - P[7]) * inv;
  planes2[4 * NC + i] = (P[0] - tr3) * inv;
  planes2[5 * NC + i] = 0.5f * (P[1] + P[3]) * inv;
  planes2[6 * NC + i] = 0.5f * (P[2] + P[6]) * inv;
  planes2[7 * NC + i] = (P[4] - tr3) * inv;
  planes2[8 * NC + i] = 0.5f * (P[5] + P[7]) * inv;
}

// ---------------------------------------------------------------------------
// Kernel 6: out = Xn + dX + dX@dX  (dX rebuilt from mixed2 compacts).
// ---------------------------------------------------------------------------
__global__ void __launch_bounds__(256) node_final_kernel(
    const float* __restrict__ Xn, const float* __restrict__ mixed2,
    float* __restrict__ out)
{
  int i = blockIdx.x * 256 + threadIdx.x;
  if (i >= NC) return;
  float cd[9];
  #pragma unroll
  for (int p = 0; p < 9; ++p) cd[p] = mixed2[p * NC + i];
  float D[9], D2[9];
  rebuild3x3(cd, D);
  #pragma unroll
  for (int r = 0; r < 3; ++r)
    #pragma unroll
    for (int c = 0; c < 3; ++c) {
      float v = 0.0f;
      #pragma unroll
      for (int k = 0; k < 3; ++k) v += D[r * 3 + k] * D[k * 3 + c];
      D2[r * 3 + c] = v;
    }
  #pragma unroll
  for (int j = 0; j < 9; ++j)
    out[(size_t)i * 9 + j] = Xn[(size_t)i * 9 + j] + D[j] + D2[j];
}

// ---------------------------------------------------------------------------
extern "C" void kernel_launch(void* const* d_in, const int* in_sizes, int n_in,
                              void* d_out, int out_size, void* d_ws, size_t ws_size,
                              hipStream_t stream) {
  (void)in_sizes; (void)n_in; (void)out_size; (void)ws_size;
  const float* X    = (const float*)d_in[0];
  const int*   ei   = (const int*)d_in[1];
  const float* ew   = (const float*)d_in[2];
  const float* ea   = (const float*)d_in[3];
  const float* W1   = (const float*)d_in[4];
  const float* b1   = (const float*)d_in[5];
  const float* W2   = (const float*)d_in[6];
  const float* b2   = (const float*)d_in[7];
  const float* W3   = (const float*)d_in[8];
  const float* b3   = (const float*)d_in[9];
  const float* Wt   = (const float*)d_in[10];

  char* ws = (char*)d_ws;
  size_t off = 0;
  _Float16* Wh = (_Float16*)(ws + off);            off += 131072;                // 59392 halfs, padded
  _Float16* afac = (_Float16*)(ws + off);          off += (size_t)EE * 192 * 2;  // 61.44 MB
  float* Xn      = (float*)(ws + off);             off += (size_t)NC * 9 * 4;    // 23.04 MB
  float* planes1 = (float*)(ws + off);             off += (size_t)NC * 9 * 4;
  float* mixed1  = (float*)(ws + off);             off += (size_t)NC * 9 * 4;
  float* msg     = (float*)(ws + off);             off += (size_t)NC * 9 * 4;
  float* planes2 = (float*)(ws + off);             off += (size_t)NC * 9 * 4;
  float* mixed2  = (float*)(ws + off);             off += (size_t)NC * 9 * 4;

  const _Float16* W1h = Wh;
  const _Float16* W2h = Wh + 2048;
  const _Float16* W3h = Wh + 10240;
  const _Float16* WtH = Wh + 34816;                // Wt[0..5], 4096 halfs each

  convert_weights_kernel<<<232, 256, 0, stream>>>(W1, W2, W3, Wt, Wh);
  hipMemsetAsync(msg, 0, (size_t)NC * 9 * 4, stream);

  edge_mlp_kernel<<<EE / 128, 256, 0, stream>>>(ea, ew, b1, b2, b3, W1h, W2h, W3h, afac);
  node_decompose_kernel<<<(NC + 255) / 256, 256, 0, stream>>>(X, Xn, planes1);
  mix_gemm_kernel<<<(9 * 625 * 4 + 7) / 8, 256, 0, stream>>>(planes1, WtH, mixed1);
  edge_scatter_kernel<<<(EE * CC + 255) / 256, 256, 0, stream>>>(ei, afac, mixed1, msg);
  node_update_kernel<<<(NC + 255) / 256, 256, 0, stream>>>(mixed1, msg, planes2);
  mix_gemm_kernel<<<(9 * 625 * 4 + 7) / 8, 256, 0, stream>>>(planes2, WtH + 3 * 4096, mixed2);
  node_final_kernel<<<(NC + 255) / 256, 256, 0, stream>>>(Xn, mixed2, (float*)d_out);
}
